// MultiHeadAttention_51238959841801
// MI455X (gfx1250) — compile-verified, matrix operand order checked by reading, not run on hardware
//
#include <hip/hip_runtime.h>

// MHA forward for B=2, L=2048, D=1024, H=16, DK=64 on gfx1250 (wave32).
// bf16 WMMA (v_wmma_f32_16x16x32_bf16) for all GEMMs, fp32 accumulation,
// flash-attention streaming softmax, async global->LDS staging of K/V tiles.

typedef __bf16 bf16x16 __attribute__((ext_vector_type(16)));
typedef float f32x8 __attribute__((ext_vector_type(8)));
typedef int v4i __attribute__((vector_size(16)));
typedef v4i __attribute__((address_space(1))) *gbl_v4i_ptr;
typedef v4i __attribute__((address_space(3))) *lds_v4i_ptr;

static constexpr int BATCH = 2;
static constexpr int SEQ   = 2048;
static constexpr int DMODEL= 1024;
static constexpr int NHEAD = 16;
static constexpr int DK    = 64;

#if __has_builtin(__builtin_amdgcn_global_load_async_to_lds_b128)
#define HAS_ASYNC_LDS 1
#else
#define HAS_ASYNC_LDS 0
#endif

// ---- helpers -------------------------------------------------------------

__device__ __forceinline__ unsigned short f2bf(float f) {
    unsigned u = __float_as_uint(f);
    unsigned r = u + 0x7FFFu + ((u >> 16) & 1u);   // round-to-nearest-even
    return (unsigned short)(r >> 16);
}

// Load one 16x32 bf16 WMMA fragment (A layout; also used for B with the
// source stored N-major so rows are contiguous in K).
// Per ISA 7.12.2: lane holds row (lane&15); K halves by (lane>>4)*8,
// elements 0..7 = K g..g+7, elements 8..15 = K 16+g..16+g+7.
__device__ __forceinline__ bf16x16 load_frag(const unsigned short* tile,
                                             int stride, int lane) {
    const int m = lane & 15;
    const int g = (lane >> 4) * 8;
    const unsigned short* p = tile + (size_t)m * stride + g;
    union { uint4 q[2]; bf16x16 v; } u;
    u.q[0] = *reinterpret_cast<const uint4*>(p);
    u.q[1] = *reinterpret_cast<const uint4*>(p + 16);
    return u.v;
}

__device__ __forceinline__ f32x8 wmma_bf16(bf16x16 a, bf16x16 b, f32x8 c) {
    return __builtin_amdgcn_wmma_f32_16x16x32_bf16(
        false, a, false, b, (short)0, c, false, false);
}

// 16-byte global -> LDS copy: async on gfx1250 toolchains that expose the
// builtin (GLOBAL_LOAD_ASYNC_TO_LDS_B128, tracked by ASYNCcnt), else a plain
// bounce through VGPRs.
__device__ __forceinline__ void async_copy16(const unsigned short* g,
                                             unsigned short* l) {
#if HAS_ASYNC_LDS
    __builtin_amdgcn_global_load_async_to_lds_b128(
        (gbl_v4i_ptr)(unsigned short*)g, (lds_v4i_ptr)l, 0, 0);
#else
    *reinterpret_cast<uint4*>(l) = *reinterpret_cast<const uint4*>(g);
#endif
}

__device__ __forceinline__ void wait_async0() {
#if HAS_ASYNC_LDS
#if __has_builtin(__builtin_amdgcn_s_wait_asynccnt)
    __builtin_amdgcn_s_wait_asynccnt(0);
#else
    asm volatile("s_wait_asynccnt 0" ::: "memory");
#endif
#endif
}

// ---- conversion kernels --------------------------------------------------

__global__ void cvt_flat(const float* __restrict__ src,
                         unsigned short* __restrict__ dst, int n) {
    int i = blockIdx.x * blockDim.x + threadIdx.x;
    if (i < n) dst[i] = f2bf(src[i]);
}

// dst[c][r] = bf16(src[r][c]) : store weights N-major for contiguous B frags.
__global__ void cvt_transpose(const float* __restrict__ src,
                              unsigned short* __restrict__ dst, int R, int C) {
    int i = blockIdx.x * blockDim.x + threadIdx.x;
    if (i < R * C) {
        int r = i / C, c = i % C;
        dst[(size_t)c * R + r] = f2bf(src[i]);
    }
}

// ---- bf16 WMMA GEMM:  out = A(MxK) @ Bt(NxK)^T + bias --------------------
// block = 128 threads (4 waves), workgroup tile 128x64, wave tile 32x64:
// two A fragments share four B fragments -> 8 WMMAs per 12 b128 loads.
// mode 0: store Q layout (B,H,L,DK) bf16
// mode 1: store K layout (B,H,L,DK) bf16
// mode 2: store V transposed (B,H,DK,L) bf16
// mode 3: store fp32 row-major MxN (final projection)
__global__ __launch_bounds__(128)
void gemm_bf16_wmma(const unsigned short* __restrict__ A,
                    const unsigned short* __restrict__ Bt,
                    const float* __restrict__ bias,
                    unsigned short* __restrict__ outb,
                    float* __restrict__ outf,
                    int M, int N, int K, int mode) {
    const int lane = threadIdx.x & 31;
    const int wave = threadIdx.x >> 5;
    const int n0 = blockIdx.x * 64;
    const int m0 = blockIdx.y * 128 + wave * 32;

    const unsigned short* Arow0 = A + (size_t)m0 * K;
    const unsigned short* Arow1 = A + (size_t)(m0 + 16) * K;
    f32x8 acc[2][4] = {};

    for (int k = 0; k < K; k += 32) {
        __builtin_prefetch(Arow0 + k + 128, 0, 0);   // global_prefetch_b8
        __builtin_prefetch(Arow1 + k + 128, 0, 0);
        bf16x16 a0 = load_frag(Arow0 + k, K, lane);
        bf16x16 a1 = load_frag(Arow1 + k, K, lane);
#pragma unroll
        for (int t = 0; t < 4; ++t) {
            bf16x16 b = load_frag(Bt + (size_t)(n0 + t * 16) * K + k, K, lane);
            acc[0][t] = wmma_bf16(a0, b, acc[0][t]);
            acc[1][t] = wmma_bf16(a1, b, acc[1][t]);
        }
    }

    const int nl = lane & 15;
    const int mh = (lane >> 4) * 8;
#pragma unroll
    for (int g = 0; g < 2; ++g) {
#pragma unroll
        for (int t = 0; t < 4; ++t) {
            const int n = n0 + t * 16 + nl;
            const float bv = bias[n];
#pragma unroll
            for (int r = 0; r < 8; ++r) {
                const float v = acc[g][t][r] + bv;
                const int m = m0 + g * 16 + mh + r;
                if (mode == 3) {
                    outf[(size_t)m * N + n] = v;
                } else {
                    const int b = m / SEQ, l = m % SEQ;
                    const int h = n >> 6, dk = n & 63;
                    if (mode == 2)   // V^T: (B,H,DK,L)
                        outb[(((size_t)b * NHEAD + h) * DK + dk) * SEQ + l] = f2bf(v);
                    else             // Q/K: (B,H,L,DK)
                        outb[(((size_t)b * NHEAD + h) * SEQ + l) * DK + dk] = f2bf(v);
                }
            }
        }
    }
}

// ---- flash attention -----------------------------------------------------
// grid (SEQ/64, NHEAD, BATCH), block 128 (4 waves). Wave owns 16 queries.
// K/V tiles for each 32-key block are staged once per workgroup into LDS via
// async global->LDS copies; all 4 waves consume them with ds_load fragments.
__global__ __launch_bounds__(128)
void flash_fwd(const unsigned short* __restrict__ Qb,
               const unsigned short* __restrict__ Kb,
               const unsigned short* __restrict__ Vtb,
               unsigned short* __restrict__ Ob) {
    __shared__ unsigned short Ktile[32][64];     // 32 keys x DK          (4 KB)
    __shared__ unsigned short Vtile[64][32];     // DK x 32 keys          (4 KB)
    __shared__ unsigned short pl[4][16 * 32];    // per-wave P bounce     (4 KB)

    const int tid  = threadIdx.x;
    const int lane = tid & 31;
    const int wave = tid >> 5;
    const int h = blockIdx.y, b = blockIdx.z;
    const int q0 = blockIdx.x * 64 + wave * 16;

    const size_t head = ((size_t)b * NHEAD + h) * (size_t)SEQ * DK;
    const unsigned short* Qh = Qb + head;
    const unsigned short* Kh = Kb + head;
    const unsigned short* Vh = Vtb + head;   // (DK, SEQ)

    // Q fragments stay resident: two K-steps cover DK=64.
    const bf16x16 qa0 = load_frag(Qh + (size_t)q0 * DK + 0,  DK, lane);
    const bf16x16 qa1 = load_frag(Qh + (size_t)q0 * DK + 32, DK, lane);

    f32x8 o0 = {}, o1 = {}, o2 = {}, o3 = {};
    float mst[8], lst[8];
#pragma unroll
    for (int r = 0; r < 8; ++r) { mst[r] = -1e30f; lst[r] = 0.0f; }
    const float scale = 0.125f;   // 1/sqrt(DK)

    unsigned short* myP = &pl[wave][0];

    for (int kb = 0; kb < SEQ; kb += 32) {
        // Cooperative async stage of K (32x64) and V^T (64x32) tiles.
        // 256 16-byte chunks each; 128 threads -> 2 chunks per tile per thread.
#pragma unroll
        for (int i = 0; i < 2; ++i) {
            const int c = tid + i * 128;
            const int kr = c >> 3, kc = (c & 7) * 8;     // K tile chunk
            async_copy16(Kh + (size_t)(kb + kr) * DK + kc, &Ktile[kr][kc]);
            const int vd = c >> 2, vc = (c & 3) * 8;     // V tile chunk
            async_copy16(Vh + (size_t)vd * SEQ + kb + vc, &Vtile[vd][vc]);
        }
        wait_async0();
        __syncthreads();

        // S = Q @ K^T for 32 keys (two 16-column C tiles), fragments from LDS.
        f32x8 s0 = {}, s1 = {};
        {
            bf16x16 k0a = load_frag(&Ktile[0][0],   64, lane);
            bf16x16 k0b = load_frag(&Ktile[0][32],  64, lane);
            s0 = wmma_bf16(qa0, k0a, s0);
            s0 = wmma_bf16(qa1, k0b, s0);
            bf16x16 k1a = load_frag(&Ktile[16][0],  64, lane);
            bf16x16 k1b = load_frag(&Ktile[16][32], 64, lane);
            s1 = wmma_bf16(qa0, k1a, s1);
            s1 = wmma_bf16(qa1, k1b, s1);
        }

        // Online softmax. C layout: lanes 0-15 hold rows 0-7 (vgpr r),
        // lanes 16-31 rows 8-15; columns live across the 16-lane half.
        float corr[8];
#pragma unroll
        for (int r = 0; r < 8; ++r) {
            float a = s0[r] * scale, c2 = s1[r] * scale;
            s0[r] = a; s1[r] = c2;
            float rm = fmaxf(a, c2);
            rm = fmaxf(rm, __shfl_xor(rm, 1, 32));
            rm = fmaxf(rm, __shfl_xor(rm, 2, 32));
            rm = fmaxf(rm, __shfl_xor(rm, 4, 32));
            rm = fmaxf(rm, __shfl_xor(rm, 8, 32));
            float mnew = fmaxf(mst[r], rm);
            float c = __expf(mst[r] - mnew);
            float p0 = __expf(s0[r] - mnew);
            float p1 = __expf(s1[r] - mnew);
            s0[r] = p0; s1[r] = p1;
            float rs = p0 + p1;
            rs += __shfl_xor(rs, 1, 32);
            rs += __shfl_xor(rs, 2, 32);
            rs += __shfl_xor(rs, 4, 32);
            rs += __shfl_xor(rs, 8, 32);
            lst[r] = lst[r] * c + rs;
            mst[r] = mnew;
            corr[r] = c;
        }
#pragma unroll
        for (int r = 0; r < 8; ++r) {
            o0[r] *= corr[r]; o1[r] *= corr[r];
            o2[r] *= corr[r]; o3[r] *= corr[r];
        }

        // C-layout -> A-layout re-tiling of P via per-wave LDS tile.
        {
            const int row = (lane >> 4) * 8;
            const int col = lane & 15;
#pragma unroll
            for (int r = 0; r < 8; ++r) {
                myP[(row + r) * 32 + col]      = f2bf(s0[r]);
                myP[(row + r) * 32 + 16 + col] = f2bf(s1[r]);
            }
        }
        __syncthreads();

        bf16x16 pa = load_frag(myP, 32, lane);
        bf16x16 v0 = load_frag(&Vtile[0][0],  32, lane);
        bf16x16 v1 = load_frag(&Vtile[16][0], 32, lane);
        bf16x16 v2 = load_frag(&Vtile[32][0], 32, lane);
        bf16x16 v3 = load_frag(&Vtile[48][0], 32, lane);
        o0 = wmma_bf16(pa, v0, o0);
        o1 = wmma_bf16(pa, v1, o1);
        o2 = wmma_bf16(pa, v2, o2);
        o3 = wmma_bf16(pa, v3, o3);
        __syncthreads();   // protect K/V/P tiles before next iteration
    }

    // Normalize and store O as bf16 in (B*L, H*DK) row-major for final GEMM.
    const int nl = lane & 15;
    const int mh = (lane >> 4) * 8;
#pragma unroll
    for (int r = 0; r < 8; ++r) {
        const float inv = 1.0f / lst[r];
        const int q = q0 + mh + r;
        const size_t rowbase = ((size_t)b * SEQ + q) * (NHEAD * DK) + (size_t)h * DK;
        Ob[rowbase +  0 + nl] = f2bf(o0[r] * inv);
        Ob[rowbase + 16 + nl] = f2bf(o1[r] * inv);
        Ob[rowbase + 32 + nl] = f2bf(o2[r] * inv);
        Ob[rowbase + 48 + nl] = f2bf(o3[r] * inv);
    }
}

// ---- host launch ---------------------------------------------------------

extern "C" void kernel_launch(void* const* d_in, const int* in_sizes, int n_in,
                              void* d_out, int out_size, void* d_ws, size_t ws_size,
                              hipStream_t stream) {
    const float* X  = (const float*)d_in[0];
    const float* WQ = (const float*)d_in[1];
    const float* bQ = (const float*)d_in[2];
    const float* WK = (const float*)d_in[3];
    const float* bK = (const float*)d_in[4];
    const float* WV = (const float*)d_in[5];
    const float* bV = (const float*)d_in[6];
    const float* WO = (const float*)d_in[7];
    const float* bO = (const float*)d_in[8];
    float* out = (float*)d_out;

    const int M = BATCH * SEQ;        // 4096
    const int N = NHEAD * DK;         // 1024
    const int K = DMODEL;             // 1024

    // Workspace layout (bytes):
    char* ws = (char*)d_ws;
    unsigned short* Xb  = (unsigned short*)(ws + 0);          // 8 MiB
    unsigned short* WQt = (unsigned short*)(ws + 8388608);    // 2 MiB each
    unsigned short* WKt = (unsigned short*)(ws + 10485760);
    unsigned short* WVt = (unsigned short*)(ws + 12582912);
    unsigned short* WOt = (unsigned short*)(ws + 14680064);
    unsigned short* Qb  = (unsigned short*)(ws + 16777216);   // 8 MiB each
    unsigned short* Kb  = (unsigned short*)(ws + 25165824);
    unsigned short* Vtb = (unsigned short*)(ws + 33554432);
    unsigned short* Ob  = (unsigned short*)(ws + 41943040);

    // 1) fp32 -> bf16 conversions (weights stored N-major).
    cvt_flat<<<(M * K + 255) / 256, 256, 0, stream>>>(X, Xb, M * K);
    cvt_transpose<<<(K * N + 255) / 256, 256, 0, stream>>>(WQ, WQt, K, N);
    cvt_transpose<<<(K * N + 255) / 256, 256, 0, stream>>>(WK, WKt, K, N);
    cvt_transpose<<<(K * N + 255) / 256, 256, 0, stream>>>(WV, WVt, K, N);
    cvt_transpose<<<(K * N + 255) / 256, 256, 0, stream>>>(WO, WOt, K, N);

    // 2) QKV projections via bf16 WMMA GEMM (128x64 macro tile).
    dim3 gg(N / 64, M / 128);
    gemm_bf16_wmma<<<gg, 128, 0, stream>>>(Xb, WQt, bQ, Qb, nullptr, M, N, K, 0);
    gemm_bf16_wmma<<<gg, 128, 0, stream>>>(Xb, WKt, bK, Kb, nullptr, M, N, K, 1);
    gemm_bf16_wmma<<<gg, 128, 0, stream>>>(Xb, WVt, bV, Vtb, nullptr, M, N, K, 2);

    // 3) Flash attention.
    flash_fwd<<<dim3(SEQ / 64, NHEAD, BATCH), 128, 0, stream>>>(Qb, Kb, Vtb, Ob);

    // 4) Output projection -> fp32.
    gemm_bf16_wmma<<<gg, 128, 0, stream>>>(Ob, WOt, bO, nullptr, out, M, N, K, 3);
}
